// SwinBlockLite_48713519071437
// MI455X (gfx1250) — matrix operand order repaired and checked by reading, not compiled, not run on hardware
//
#include <hip/hip_runtime.h>
#include <hip/hip_bf16.h>
#include <stdint.h>

// ---------------------------------------------------------------------------
// CDNA5 / gfx1250 Swin-lite block: bf16 WMMA + async global->LDS double buffer
// ---------------------------------------------------------------------------

typedef __attribute__((ext_vector_type(16))) __bf16 v16bf;
typedef __attribute__((ext_vector_type(8)))  float  v8f;

union Frag { v16bf v; uint4 u[2]; };

#define LPITCH 72   // 64 + 8 bf16 pad -> 144B row pitch, kills ds_read_b128 bank conflicts

// ---- CDNA5 async global->LDS (ASYNCcnt) -----------------------------------
__device__ __forceinline__ void async_ld_b128(const void* gsrc, const void* ldst) {
    unsigned lds_off = (unsigned)(uintptr_t)ldst;   // flat addr low 32 bits == LDS offset
    asm volatile("global_load_async_to_lds_b128 %0, %1, off"
                 :: "v"(lds_off), "v"(gsrc) : "memory");
}
__device__ __forceinline__ void wait_async0() {
    asm volatile("s_wait_asynccnt 0x0" ::: "memory");
}

__device__ __forceinline__ float red16_max(float v) {
#pragma unroll
    for (int d = 8; d >= 1; d >>= 1) v = fmaxf(v, __shfl_xor(v, d, 32));
    return v;
}
__device__ __forceinline__ float red16_sum(float v) {
#pragma unroll
    for (int d = 8; d >= 1; d >>= 1) v += __shfl_xor(v, d, 32);
    return v;
}
__device__ __forceinline__ float red32_sum(float v) {
#pragma unroll
    for (int d = 16; d >= 1; d >>= 1) v += __shfl_xor(v, d, 32);
    return v;
}

// ---------------------------------------------------------------------------
// f32 -> bf16 weight convert
// ---------------------------------------------------------------------------
__global__ __launch_bounds__(256) void cvt_f32_bf16(const float* __restrict__ src,
                                                    __bf16* __restrict__ dst, int n) {
    int i = blockIdx.x * 256 + threadIdx.x;
    if (i < n) dst[i] = (__bf16)src[i];
}

// ---------------------------------------------------------------------------
// LN1 over gathered window tokens: x(B,C,H,W) -> normed bf16 [M=131072, 512]
// token m = (win=b*256+nh*16+nw)*64 + (s=wy*8+wx)
// ---------------------------------------------------------------------------
__global__ __launch_bounds__(256) void ln1_kernel(const float* __restrict__ x,
                                                  const float* __restrict__ g,
                                                  const float* __restrict__ bt,
                                                  __bf16* __restrict__ out) {
    const int wv = threadIdx.x >> 5, lane = threadIdx.x & 31;
    const long m = (long)blockIdx.x * 8 + wv;
    const long win = m >> 6; const int s = (int)(m & 63);
    const long bI = win >> 8; const int rem = (int)(win & 255);
    const int hp = (rem >> 4) * 8 + (s >> 3);
    const int wp = (rem & 15) * 8 + (s & 7);
    const float* xp = x + (bI * 512 * 128 + hp) * 128 + wp;   // + ch*16384

    float vals[16]; float sum = 0.f;
#pragma unroll
    for (int i = 0; i < 16; ++i) {
        int ch = i * 32 + lane;
        vals[i] = xp[(long)ch * 16384];
        sum += vals[i];
    }
    sum = red32_sum(sum);
    const float mu = sum * (1.f / 512.f);
    float vv = 0.f;
#pragma unroll
    for (int i = 0; i < 16; ++i) { float d = vals[i] - mu; vv += d * d; }
    vv = red32_sum(vv);
    const float rstd = rsqrtf(vv * (1.f / 512.f) + 1e-5f);
#pragma unroll
    for (int i = 0; i < 16; ++i) {
        int ch = i * 32 + lane;
        out[m * 512 + ch] = (__bf16)((vals[i] - mu) * rstd * g[ch] + bt[ch]);
    }
}

// ---------------------------------------------------------------------------
// LN2 over contiguous token rows of y1 (f32 [M,512]) -> bf16 [M,512]
// ---------------------------------------------------------------------------
__global__ __launch_bounds__(256) void ln2_kernel(const float* __restrict__ y1,
                                                  const float* __restrict__ g,
                                                  const float* __restrict__ bt,
                                                  __bf16* __restrict__ out) {
    const int wv = threadIdx.x >> 5, lane = threadIdx.x & 31;
    const long m = (long)blockIdx.x * 8 + wv;
    const float* yp = y1 + m * 512;

    float vals[16]; float sum = 0.f;
#pragma unroll
    for (int i = 0; i < 16; ++i) {
        int ch = i * 32 + lane;
        vals[i] = yp[ch];
        sum += vals[i];
    }
    sum = red32_sum(sum);
    const float mu = sum * (1.f / 512.f);
    float vv = 0.f;
#pragma unroll
    for (int i = 0; i < 16; ++i) { float d = vals[i] - mu; vv += d * d; }
    vv = red32_sum(vv);
    const float rstd = rsqrtf(vv * (1.f / 512.f) + 1e-5f);
#pragma unroll
    for (int i = 0; i < 16; ++i) {
        int ch = i * 32 + lane;
        out[m * 512 + ch] = (__bf16)((vals[i] - mu) * rstd * g[ch] + bt[ch]);
    }
}

// ---------------------------------------------------------------------------
// Tiled bf16 WMMA GEMM: D[M,N] = A[M,K] * B[N,K]^T, fused epilogues.
// 128 threads (4 waves), 64x64 tile, 64-deep K stage, async double-buffered LDS.
// ---------------------------------------------------------------------------
struct EpiArgs {
    const float* bias;
    const float* xsrc;   // EPI_OUT: residual gather source x(B,C,H,W)
    const float* y1r;    // EPI_FC2: residual read
    float*       y1w;    // EPI_OUT: write f32 [M,512]
    __bf16*      b0;     // EPI_QKV: q   | EPI_FC1: h
    __bf16*      b1;     // EPI_QKV: k
    __bf16*      b2;     // EPI_QKV: v^T
    float*       outF;   // EPI_FC2: final (B,C,H,W)
};

enum { EPI_QKV = 0, EPI_OUT = 1, EPI_FC1 = 2, EPI_FC2 = 3 };

template <int EP>
__global__ __launch_bounds__(128) void wgemm_bf16(const __bf16* __restrict__ A,
                                                  const __bf16* __restrict__ Bw,
                                                  int K, EpiArgs ea) {
    __shared__ alignas(16) __bf16 sA[2][64 * LPITCH];
    __shared__ alignas(16) __bf16 sB[2][64 * LPITCH];
    const int t = threadIdx.x;
    const int wv = t >> 5, lane = t & 31;
    const int hf = lane >> 4, ln = lane & 15;
    const long m0 = (long)blockIdx.x * 64;
    const int  n0 = blockIdx.y * 64;

    // async-stage one 64-deep K slice into LDS buffer `bi`
    auto stage = [&](int k0, int bi) {
#pragma unroll
        for (int i = 0; i < 8; ++i) {
            int flat = t + 128 * i;          // 0..1023 -> 512 x 16B for A + 512 for B
            int isB  = flat >> 9;
            int f    = flat & 511;
            int row  = f >> 3, seg = f & 7;
            const __bf16* src = isB ? (Bw + (long)(n0 + row) * K + k0 + seg * 8)
                                    : (A  + (m0 + row) * (long)K + k0 + seg * 8);
            const __bf16* dst = (isB ? sB[bi] : sA[bi]) + row * LPITCH + seg * 8;
            async_ld_b128(src, dst);
        }
    };

    v8f c[4] = {};
    int buf = 0;
    stage(0, 0);

    for (int k0 = 0; k0 < K; k0 += 64) {
        wait_async0();        // this wave's async LDS writes (current buf) complete
        __syncthreads();      // all waves' writes visible; all done computing other buf
        const int nxt = buf ^ 1;
        if (k0 + 64 < K) stage(k0 + 64, nxt);   // overlap next HBM stage with WMMAs

        const int mr = wv * 16 + ln;
#pragma unroll
        for (int kk = 0; kk < 64; kk += 32) {
            Frag a;   // A layout: lanes<16 K 0..15, lanes>=16 K 16..31 (matches B)
            a.u[0] = *(const uint4*)(sA[buf] + mr * LPITCH + kk + hf * 16);
            a.u[1] = *(const uint4*)(sA[buf] + mr * LPITCH + kk + hf * 16 + 8);
#pragma unroll
            for (int j = 0; j < 4; ++j) {
                Frag b;   // B layout: lanes<16 K 0..15, lanes>=16 K 16..31 of column (j*16+ln)
                const __bf16* bp = sB[buf] + (j * 16 + ln) * LPITCH + kk + hf * 16;
                b.u[0] = *(const uint4*)bp;
                b.u[1] = *(const uint4*)(bp + 8);
                c[j] = __builtin_amdgcn_wmma_f32_16x16x32_bf16(false, a.v, false, b.v,
                                                               (short)0, c[j], false, false);
            }
        }
        buf = nxt;
    }

    // D layout: row = m0 + wv*16 + hf*8 + p, col = n0 + j*16 + ln
    const long mb = m0 + wv * 16 + hf * 8;
#pragma unroll
    for (int j = 0; j < 4; ++j) {
        const int n = n0 + j * 16 + ln;
#pragma unroll
        for (int p = 0; p < 8; ++p) {
            const long m = mb + p;
            float v = c[j][p];
            if constexpr (EP == EPI_QKV) {
                v += ea.bias[n];
                int sec = n >> 9, ch = n & 511, head = ch >> 6, d = ch & 63;
                long win = m >> 6; int s = (int)(m & 63);
                long hb = (win * 8 + head) * 4096;
                if (sec == 0)      ea.b0[hb + s * 64 + d] = (__bf16)v;   // q [s][d]
                else if (sec == 1) ea.b1[hb + s * 64 + d] = (__bf16)v;   // k [t][d]
                else               ea.b2[hb + d * 64 + s] = (__bf16)v;   // v^T [d][t]
            } else if constexpr (EP == EPI_OUT) {
                v += ea.bias[n];
                long win = m >> 6; int s = (int)(m & 63);
                long bI = win >> 8; int rem = (int)(win & 255);
                int hp = (rem >> 4) * 8 + (s >> 3);
                int wp = (rem & 15) * 8 + (s & 7);
                float xv = ea.xsrc[((bI * 512 + n) * 128 + hp) * 128 + wp];
                ea.y1w[m * 512 + n] = v + xv;
            } else if constexpr (EP == EPI_FC1) {
                v += ea.bias[n];
                v = v / (1.f + __expf(-v));                              // SiLU
                ea.b0[m * 1024 + n] = (__bf16)v;
            } else { // EPI_FC2: +bias +residual, scatter to (B,C,H,W)
                v += ea.bias[n] + ea.y1r[m * 512 + n];
                long win = m >> 6; int s = (int)(m & 63);
                long bI = win >> 8; int rem = (int)(win & 255);
                int hp = (rem >> 4) * 8 + (s >> 3);
                int wp = (rem & 15) * 8 + (s & 7);
                ea.outF[((bI * 512 + n) * 128 + hp) * 128 + wp] = v;
            }
        }
    }
}

// ---------------------------------------------------------------------------
// Attention: one block per (window, head). scores = q k^T /8, softmax, o = P v
// q,k: [wh][s][d] bf16; vT: [wh][d][t] bf16; o -> token-major bf16 [M,512]
// ---------------------------------------------------------------------------
__global__ __launch_bounds__(128) void attn_kernel(const __bf16* __restrict__ Q,
                                                   const __bf16* __restrict__ Kb,
                                                   const __bf16* __restrict__ Vt,
                                                   __bf16* __restrict__ O) {
    __shared__ alignas(16) __bf16 sQ[64 * LPITCH];
    __shared__ alignas(16) __bf16 sK[64 * LPITCH];
    __shared__ alignas(16) __bf16 sV[64 * LPITCH];
    __shared__ alignas(16) __bf16 sP[64 * LPITCH];
    const int wh = blockIdx.x;
    const long base = (long)wh * 4096;
    const int t = threadIdx.x, wv = t >> 5, lane = t & 31;
    const int hf = lane >> 4, ln = lane & 15;

#pragma unroll
    for (int i = 0; i < 12; ++i) {
        int flat = t + 128 * i;              // 0..1535 -> 512 x 16B each of q,k,vT
        int which = flat >> 9;
        int f = flat & 511, row = f >> 3, seg = f & 7;
        const __bf16* src = (which == 0 ? Q : which == 1 ? Kb : Vt) + base + row * 64 + seg * 8;
        const __bf16* dst = (which == 0 ? sQ : which == 1 ? sK : sV) + row * LPITCH + seg * 8;
        async_ld_b128(src, dst);
    }
    wait_async0();
    __syncthreads();

    const int mr = wv * 16 + ln;
    v8f c[4] = {};
#pragma unroll
    for (int kk = 0; kk < 64; kk += 32) {
        Frag a;
        a.u[0] = *(const uint4*)(sQ + mr * LPITCH + kk + hf * 16);
        a.u[1] = *(const uint4*)(sQ + mr * LPITCH + kk + hf * 16 + 8);
#pragma unroll
        for (int j = 0; j < 4; ++j) {
            Frag b;
            const __bf16* bp = sK + (j * 16 + ln) * LPITCH + kk + hf * 16;
            b.u[0] = *(const uint4*)bp; b.u[1] = *(const uint4*)(bp + 8);
            c[j] = __builtin_amdgcn_wmma_f32_16x16x32_bf16(false, a.v, false, b.v,
                                                           (short)0, c[j], false, false);
        }
    }

    // softmax over each row (64 cols spread across 4 tiles x 16 lanes of a half-wave)
    const float scale = 0.125f;   // 1/sqrt(64)
#pragma unroll
    for (int p = 0; p < 8; ++p) {
        float mx = fmaxf(fmaxf(c[0][p], c[1][p]), fmaxf(c[2][p], c[3][p]));
        mx = red16_max(mx);
        float s = 0.f;
#pragma unroll
        for (int j = 0; j < 4; ++j) { float e = __expf((c[j][p] - mx) * scale); c[j][p] = e; s += e; }
        s = red16_sum(s);
        const float inv = 1.f / s;
        const int row = wv * 16 + hf * 8 + p;
#pragma unroll
        for (int j = 0; j < 4; ++j)
            sP[row * LPITCH + j * 16 + ln] = (__bf16)(c[j][p] * inv);
    }
    __syncthreads();

    v8f o[4] = {};
#pragma unroll
    for (int kk = 0; kk < 64; kk += 32) {
        Frag a;
        a.u[0] = *(const uint4*)(sP + mr * LPITCH + kk + hf * 16);
        a.u[1] = *(const uint4*)(sP + mr * LPITCH + kk + hf * 16 + 8);
#pragma unroll
        for (int j = 0; j < 4; ++j) {
            Frag b;
            const __bf16* bp = sV + (j * 16 + ln) * LPITCH + kk + hf * 16;
            b.u[0] = *(const uint4*)bp; b.u[1] = *(const uint4*)(bp + 8);
            o[j] = __builtin_amdgcn_wmma_f32_16x16x32_bf16(false, a.v, false, b.v,
                                                           (short)0, o[j], false, false);
        }
    }

    const long win = wh >> 3;
    const int head = wh & 7;
    const long mb = win * 64 + wv * 16 + hf * 8;
#pragma unroll
    for (int j = 0; j < 4; ++j) {
        const int ch = head * 64 + j * 16 + ln;
#pragma unroll
        for (int p = 0; p < 8; ++p)
            O[(mb + p) * 512 + ch] = (__bf16)(o[j][p]);
    }
}

// ---------------------------------------------------------------------------
// Host launcher
// ---------------------------------------------------------------------------
extern "C" void kernel_launch(void* const* d_in, const int* in_sizes, int n_in,
                              void* d_out, int out_size, void* d_ws, size_t ws_size,
                              hipStream_t stream) {
    (void)in_sizes; (void)n_in; (void)out_size; (void)ws_size;
    const float* x     = (const float*)d_in[0];
    const float* ln1_g = (const float*)d_in[1];
    const float* ln1_b = (const float*)d_in[2];
    const float* in_w  = (const float*)d_in[3];
    const float* in_b  = (const float*)d_in[4];
    const float* out_w = (const float*)d_in[5];
    const float* out_b = (const float*)d_in[6];
    const float* ln2_g = (const float*)d_in[7];
    const float* ln2_b = (const float*)d_in[8];
    const float* fc1_w = (const float*)d_in[9];
    const float* fc1_b = (const float*)d_in[10];
    const float* fc2_w = (const float*)d_in[11];
    const float* fc2_b = (const float*)d_in[12];
    float* out = (float*)d_out;

    const long   M    = 8L * 128 * 128;            // 131072 tokens
    const size_t SZMC = (size_t)M * 512 * 2;       // 128 MB bf16 [M,512]

    char* ws = (char*)d_ws;
    size_t off = 0;
    auto take = [&](size_t bytes) { size_t o = off; off += (bytes + 255) & ~(size_t)255; return o; };

    __bf16* w_in  = (__bf16*)(ws + take((size_t)1536 * 512 * 2));
    __bf16* w_out = (__bf16*)(ws + take((size_t)512 * 512 * 2));
    __bf16* w_fc1 = (__bf16*)(ws + take((size_t)1024 * 512 * 2));
    __bf16* w_fc2 = (__bf16*)(ws + take((size_t)512 * 1024 * 2));
    __bf16* nrm1  = (__bf16*)(ws + take(SZMC));
    __bf16* qb    = (__bf16*)(ws + take(SZMC));
    __bf16* kb    = (__bf16*)(ws + take(SZMC));
    __bf16* vtb   = (__bf16*)(ws + take(SZMC));
    float*  y1    = (float*)(ws + take((size_t)M * 512 * 4));
    // reuse (strictly sequential kernel dependencies on the same stream):
    __bf16* obuf = nrm1;          // normed1 dead after QKV GEMM
    __bf16* nrm2 = qb;            // q dead after attention
    __bf16* hbuf = kb;            // k+vT (contiguous 256 MB) dead after attention

    // 1) weights -> bf16
    cvt_f32_bf16<<<(1536 * 512 + 255) / 256, 256, 0, stream>>>(in_w,  w_in,  1536 * 512);
    cvt_f32_bf16<<<(512  * 512 + 255) / 256, 256, 0, stream>>>(out_w, w_out, 512  * 512);
    cvt_f32_bf16<<<(1024 * 512 + 255) / 256, 256, 0, stream>>>(fc1_w, w_fc1, 1024 * 512);
    cvt_f32_bf16<<<(512 * 1024 + 255) / 256, 256, 0, stream>>>(fc2_w, w_fc2, 512 * 1024);

    // 2) window-gather + LN1
    ln1_kernel<<<(int)(M / 8), 256, 0, stream>>>(x, ln1_g, ln1_b, nrm1);

    // 3) QKV projection (M x 1536, K=512), scatter into q/k/vT head layouts
    { EpiArgs e{}; e.bias = in_b; e.b0 = qb; e.b1 = kb; e.b2 = vtb;
      wgemm_bf16<EPI_QKV><<<dim3(2048, 24), 128, 0, stream>>>(nrm1, w_in, 512, e); }

    // 4) windowed attention, one block per (window, head)
    attn_kernel<<<2048 * 8, 128, 0, stream>>>(qb, kb, vtb, obuf);

    // 5) output projection + bias + residual(x gather) -> y1 f32
    { EpiArgs e{}; e.bias = out_b; e.xsrc = x; e.y1w = y1;
      wgemm_bf16<EPI_OUT><<<dim3(2048, 8), 128, 0, stream>>>(obuf, w_out, 512, e); }

    // 6) LN2
    ln2_kernel<<<(int)(M / 8), 256, 0, stream>>>(y1, ln2_g, ln2_b, nrm2);

    // 7) fc1 + SiLU
    { EpiArgs e{}; e.bias = fc1_b; e.b0 = hbuf;
      wgemm_bf16<EPI_FC1><<<dim3(2048, 16), 128, 0, stream>>>(nrm2, w_fc1, 512, e); }

    // 8) fc2 + bias + residual + window-merge scatter to (B,C,H,W)
    { EpiArgs e{}; e.bias = fc2_b; e.y1r = y1; e.outF = out;
      wgemm_bf16<EPI_FC2><<<dim3(2048, 8), 128, 0, stream>>>(hbuf, w_fc2, 1024, e); }
}
